// MultiHeadAttention_3977139716855
// MI455X (gfx1250) — matrix operand order repaired and checked by reading, not compiled, not run on hardware
//
#include <hip/hip_runtime.h>
#include <hip/hip_bf16.h>

// ---------------------------------------------------------------------------
// MHA forward, B=4, S=2048, D=1024, H=16, Dk=64 on gfx1250 (CDNA5).
// bf16 WMMA (v_wmma_f32_16x16x32_bf16), fp32 accumulate, flash attention.
// Double-buffered tiles staged with GLOBAL_LOAD_ASYNC_TO_LDS_B128 (ASYNCcnt)
// when available; transposes/conversions hoisted into one-time prepasses.
//
// Fragment layouts (ISA 7.12.2):
//   A (16x32, 16-bit): lane holds row m = lane%16; elements i=0..15 map to
//     k = (i/8)*16 + (lane>=16)*8 + i%8  -> two contiguous 16B chunks.
//   B (32x16, 16-bit): lane holds col n = lane%16; elements i map to k as for A.
//   C/D (16x16 f32):  VGPR r, lane l -> row (l<16 ? r : r+8), col l%16.
// ---------------------------------------------------------------------------

typedef __bf16 bf16_t;
typedef __attribute__((ext_vector_type(16))) __bf16 v16bf;
typedef __attribute__((ext_vector_type(8)))  __bf16 v8bf;
typedef __attribute__((ext_vector_type(4)))  __bf16 v4bf;
typedef __attribute__((ext_vector_type(8)))  float  v8f;
typedef __attribute__((ext_vector_type(4)))  float  v4f;
typedef __attribute__((ext_vector_type(4)))  int    v4i;

typedef __attribute__((address_space(1))) v4i gv4i;   // global int4
typedef __attribute__((address_space(3))) v4i lv4i;   // LDS int4

#define D_MODEL 1024
#define N_HEADS 16
#define D_K     64
#define BATCH   4
#define SEQ     2048
#define ROWS    (BATCH * SEQ)   // 8192

#if __has_builtin(__builtin_amdgcn_global_load_async_to_lds_b128) && \
    __has_builtin(__builtin_amdgcn_s_wait_asynccnt)
#define HAVE_ASYNC_LDS 1
#endif

// 16-byte tile-staging copy: async DMA to LDS when available, else sync.
__device__ __forceinline__ void stage16(bf16_t* lds_dst, const bf16_t* gsrc) {
#ifdef HAVE_ASYNC_LDS
    __builtin_amdgcn_global_load_async_to_lds_b128(
        (gv4i*)(v4i*)const_cast<bf16_t*>(gsrc),
        (lv4i*)(v4i*)lds_dst, 0, 0);
#else
    *(v8bf*)lds_dst = *(const v8bf*)gsrc;
#endif
}

__device__ __forceinline__ void stage_wait() {
#ifdef HAVE_ASYNC_LDS
    __builtin_amdgcn_s_wait_asynccnt(0);
#endif
}

__device__ __forceinline__ v8f wmma_bf16(v16bf a, v16bf b, v8f c) {
    return __builtin_amdgcn_wmma_f32_16x16x32_bf16(
        false, a, false, b, (short)0, c, false, false);
}

__device__ __forceinline__ v16bf load_frag(const bf16_t* lo, const bf16_t* hi) {
    v8bf a = *(const v8bf*)lo;
    v8bf b = *(const v8bf*)hi;
    v16bf r;
#pragma unroll
    for (int i = 0; i < 8; ++i) { r[i] = a[i]; r[8 + i] = b[i]; }
    return r;
}

__device__ __forceinline__ v16bf load_bcol(const bf16_t* base, int stride, int hi) {
    v16bf r;
#pragma unroll
    for (int i = 0; i < 8; ++i) {
        r[i] = base[(hi * 8 + i) * stride];
        r[8 + i] = base[(16 + hi * 8 + i) * stride];
    }
    return r;
}

__device__ __forceinline__ float half_max(float v) {
#pragma unroll
    for (int m = 8; m >= 1; m >>= 1) v = fmaxf(v, __shfl_xor(v, m, 32));
    return v;
}
__device__ __forceinline__ float half_sum(float v) {
#pragma unroll
    for (int m = 8; m >= 1; m >>= 1) v += __shfl_xor(v, m, 32);
    return v;
}

// ---------------------------------------------------------------------------
// Prepass 1: x (fp32) -> xb (bf16), coalesced.
// ---------------------------------------------------------------------------
__global__ __launch_bounds__(256) void convert_x_kernel(
    const float* __restrict__ x, bf16_t* __restrict__ xb)
{
    size_t i4 = ((size_t)blockIdx.x * 256 + threadIdx.x) * 4;
    v4f v = *(const v4f*)(x + i4);
    v4bf o;
#pragma unroll
    for (int j = 0; j < 4; ++j) o[j] = (bf16_t)v[j];
    *(v4bf*)(xb + i4) = o;
}

// ---------------------------------------------------------------------------
// Prepass 2: W[n][k] (fp32) -> Wt[k][n] (bf16), LDS-tiled transpose.
// ---------------------------------------------------------------------------
__global__ __launch_bounds__(128) void transpose_w_kernel(
    const float* __restrict__ Wq, const float* __restrict__ Wk,
    const float* __restrict__ Wv, const float* __restrict__ Wo,
    bf16_t* __restrict__ Wt)
{
    __shared__ __align__(16) bf16_t Ts[64][64];
    const int t = threadIdx.x;
    const int k0 = blockIdx.x * 64, n0 = blockIdx.y * 64, z = blockIdx.z;
    const float* W = (z == 0) ? Wq : (z == 1) ? Wk : (z == 2) ? Wv : Wo;
    bf16_t* dst = Wt + (size_t)z * D_MODEL * D_MODEL;

#pragma unroll
    for (int i = 0; i < 8; ++i) {
        int u = t + i * 128;
        int row = u >> 4, c4 = (u & 15) * 4;
        v4f w = *(const v4f*)(W + (size_t)(n0 + row) * D_MODEL + k0 + c4);
#pragma unroll
        for (int j = 0; j < 4; ++j) Ts[c4 + j][row] = (bf16_t)w[j];
    }
    __syncthreads();
#pragma unroll
    for (int i = 0; i < 4; ++i) {
        int u = t + i * 128;
        int kr = u >> 3, c8 = (u & 7) * 8;
        *(v8bf*)(dst + (size_t)(k0 + kr) * D_MODEL + n0 + c8) =
            *(const v8bf*)&Ts[kr][c8];
    }
}

// ---------------------------------------------------------------------------
// Kernel 1: fused Q/K/V projection, double-buffered async staging.
// grid = (ROWS/128, D/64, 3); block 128 (4 waves, 32x64 per wave).
// ---------------------------------------------------------------------------
__global__ __launch_bounds__(128) void qkv_proj_kernel(
    const bf16_t* __restrict__ xb, const bf16_t* __restrict__ Wt,
    bf16_t* __restrict__ Qb, bf16_t* __restrict__ Kt, bf16_t* __restrict__ Vb)
{
    __shared__ __align__(32) bf16_t As[2][128][32];   // [m][k]
    __shared__ __align__(32) bf16_t Bs[2][32][64];    // [k][n]

    const int t = threadIdx.x;
    const int wave = t >> 5, lane = t & 31, l16 = lane & 15, hi = lane >> 4;
    const int m0 = blockIdx.x * 128;
    const int n0 = blockIdx.y * 64;
    const int z  = blockIdx.z;
    const bf16_t* wtz = Wt + (size_t)z * D_MODEL * D_MODEL;

    auto stage = [&](int buf, int kkc) {
#pragma unroll
        for (int i = 0; i < 4; ++i) {               // A: 512 v8bf units
            int u = t + i * 128;
            int row = u >> 2, c8 = (u & 3) * 8;
            stage16(&As[buf][row][c8],
                    xb + (size_t)(m0 + row) * D_MODEL + kkc + c8);
        }
#pragma unroll
        for (int i = 0; i < 2; ++i) {               // B: 256 v8bf units
            int u = t + i * 128;
            int kr = u >> 3, c8 = (u & 7) * 8;
            stage16(&Bs[buf][kr][c8],
                    wtz + (size_t)(kkc + kr) * D_MODEL + n0 + c8);
        }
    };

    v8f acc[2][4];
#pragma unroll
    for (int ms = 0; ms < 2; ++ms)
#pragma unroll
        for (int n = 0; n < 4; ++n)
#pragma unroll
            for (int r = 0; r < 8; ++r) acc[ms][n][r] = 0.0f;

    stage(0, 0);
    stage_wait();
    __syncthreads();

    for (int kk = 0; kk < D_MODEL; kk += 32) {
        const int cur = (kk >> 5) & 1;
        if (kk + 32 < D_MODEL) stage(cur ^ 1, kk + 32);  // overlap with WMMA

        v16bf a[2];
#pragma unroll
        for (int ms = 0; ms < 2; ++ms) {
            const bf16_t* ar = &As[cur][wave * 32 + ms * 16 + l16][0];
            a[ms] = load_frag(ar + hi * 8, ar + 16 + hi * 8);
        }
#pragma unroll
        for (int n = 0; n < 4; ++n) {
            v16bf b = load_bcol(&Bs[cur][0][n * 16 + l16], 64, hi);  // lane%16 = n col
#pragma unroll
            for (int ms = 0; ms < 2; ++ms)
                acc[ms][n] = wmma_bf16(a[ms], b, acc[ms][n]);
        }
        stage_wait();
        __syncthreads();
    }

    if (z == 1) {
        // K: store transposed (bh, dk, s); one b128 store per subtile.
#pragma unroll
        for (int ms = 0; ms < 2; ++ms)
#pragma unroll
            for (int n = 0; n < 4; ++n) {
                int e  = n0 + n * 16 + l16;
                int hh = e >> 6, dk = e & 63;
                int sg0 = m0 + wave * 32 + ms * 16 + (hi ? 8 : 0);
                int bb = sg0 >> 11, ss0 = sg0 & (SEQ - 1);
                v8bf kv;
#pragma unroll
                for (int r = 0; r < 8; ++r) kv[r] = (bf16_t)acc[ms][n][r];
                *(v8bf*)(Kt + (((size_t)(bb * N_HEADS + hh)) * D_K + dk) * SEQ
                             + ss0) = kv;
            }
    } else {
        bf16_t* Out = (z == 0) ? Qb : Vb;
        const float scale = (z == 0) ? 0.125f : 1.0f;   // 1/sqrt(64) into Q
#pragma unroll
        for (int ms = 0; ms < 2; ++ms)
#pragma unroll
            for (int n = 0; n < 4; ++n) {
                int e  = n0 + n * 16 + l16;
                int hh = e >> 6, dk = e & 63;
#pragma unroll
                for (int r = 0; r < 8; ++r) {
                    int sg = m0 + wave * 32 + ms * 16 + (hi ? 8 : 0) + r;
                    int bb = sg >> 11, ss = sg & (SEQ - 1);
                    Out[(((size_t)bb * N_HEADS + hh) * SEQ + ss) * D_K + dk] =
                        (bf16_t)(acc[ms][n][r] * scale);
                }
            }
    }
}

// ---------------------------------------------------------------------------
// Kernel 2: flash attention, double-buffered async K/V staging.
// grid = (S/64, B*H); block 128 (4 waves, 16 queries each), 64-key tiles.
// ---------------------------------------------------------------------------
__global__ __launch_bounds__(128) void flash_attn_kernel(
    const bf16_t* __restrict__ Qb, const bf16_t* __restrict__ Kt,
    const bf16_t* __restrict__ Vb, const int* __restrict__ mask,
    bf16_t* __restrict__ ctx)
{
    __shared__ __align__(32) bf16_t Kts[2][64][64];   // [d][key]
    __shared__ __align__(32) bf16_t Vs[2][64][64];    // [key][d]
    __shared__ __align__(32) bf16_t Pb[4][16][64];    // per-wave P relayout

    const int t = threadIdx.x;
    const int wave = t >> 5, lane = t & 31, l16 = lane & 15, hi = lane >> 4;
    const int bh = blockIdx.y;
    const int b = bh >> 4, h = bh & 15;
    const int q0 = blockIdx.x * 64;
    const int row_off = wave * 16 + (hi ? 8 : 0);

    const bf16_t* ktp = Kt + (size_t)bh * D_K * SEQ;   // [d][s]
    const bf16_t* vbp = Vb + (size_t)bh * SEQ * D_K;   // [s][d]

    auto stage = [&](int buf, int kb) {
#pragma unroll
        for (int i = 0; i < 4; ++i) {
            int u = t + i * 128;
            int row = u >> 3, c8 = (u & 7) * 8;
            stage16(&Kts[buf][row][c8], ktp + (size_t)row * SEQ + kb + c8);
            stage16(&Vs[buf][row][c8],  vbp + (size_t)(kb + row) * D_K + c8);
        }
    };

    const int qrow = q0 + wave * 16 + l16;
    const bf16_t* qp = Qb + ((size_t)bh * SEQ + qrow) * D_K;
    v16bf a_q[2];
    a_q[0] = load_frag(qp + hi * 8,      qp + 16 + hi * 8);
    a_q[1] = load_frag(qp + 32 + hi * 8, qp + 48 + hi * 8);

    float m_run[8], l_run[8];
    v8f o_acc[4];
#pragma unroll
    for (int r = 0; r < 8; ++r) { m_run[r] = -1e30f; l_run[r] = 0.0f; }
#pragma unroll
    for (int n = 0; n < 4; ++n)
#pragma unroll
        for (int r = 0; r < 8; ++r) o_acc[n][r] = 0.0f;

    const int nkt = q0 / 64 + 1;   // causal: diagonal blocks inclusive
    stage(0, 0);
    stage_wait();
    __syncthreads();

    for (int kt = 0; kt < nkt; ++kt) {
        const int kb = kt * 64;
        const int cur = kt & 1;
        if (kt + 1 < nkt) stage(cur ^ 1, kb + 64);   // overlap with compute

        // Scores for four 16-key groups; K-dim = d = 64 -> 2 WMMA steps.
        v8f s_acc[4];
#pragma unroll
        for (int kn = 0; kn < 4; ++kn) {
            v16bf b0 = load_bcol(&Kts[cur][0][kn * 16 + l16], 64, hi);
            v16bf b1 = load_bcol(&Kts[cur][32][kn * 16 + l16], 64, hi);
            v8f s;
#pragma unroll
            for (int r = 0; r < 8; ++r) s[r] = 0.0f;
            s = wmma_bf16(a_q[0], b0, s);
            s = wmma_bf16(a_q[1], b1, s);
            s_acc[kn] = s;
        }

        // Causal + padding mask (C layout).
#pragma unroll
        for (int kn = 0; kn < 4; ++kn) {
            int kcol = kb + kn * 16 + l16;
            bool padz = (mask[b * SEQ + kcol] == 0);
#pragma unroll
            for (int r = 0; r < 8; ++r) {
                int qg = q0 + row_off + r;
                if (padz || kcol > qg) s_acc[kn][r] = -1e9f;
            }
        }

        // Online softmax: combine subtiles in-lane, one butterfly per row.
#pragma unroll
        for (int r = 0; r < 8; ++r) {
            float mt = s_acc[0][r];
#pragma unroll
            for (int kn = 1; kn < 4; ++kn) mt = fmaxf(mt, s_acc[kn][r]);
            mt = half_max(mt);
            float mn = fmaxf(m_run[r], mt);
            float sc = __expf(m_run[r] - mn);
            float psum = 0.0f;
            int prow = (hi ? 8 : 0) + r;
#pragma unroll
            for (int kn = 0; kn < 4; ++kn) {
                float p = __expf(s_acc[kn][r] - mn);
                psum += p;
                Pb[wave][prow][kn * 16 + l16] = (bf16_t)p;
            }
            l_run[r] = l_run[r] * sc + half_sum(psum);
            m_run[r] = mn;
#pragma unroll
            for (int n = 0; n < 4; ++n) o_acc[n][r] *= sc;
        }

        // P as A fragments (two 32-key steps) via per-wave LDS (in-order ds).
        const bf16_t* pp = &Pb[wave][l16][0];
        v16bf ap0 = load_frag(pp + hi * 8,      pp + 16 + hi * 8);
        v16bf ap1 = load_frag(pp + 32 + hi * 8, pp + 48 + hi * 8);

        // ctx += P @ V : B fragment lane = dk column, elements = 16 keys.
#pragma unroll
        for (int n = 0; n < 4; ++n) {
            v16bf bv0 = load_bcol(&Vs[cur][0][n * 16 + l16], 64, hi);
            v16bf bv1 = load_bcol(&Vs[cur][32][n * 16 + l16], 64, hi);
            o_acc[n] = wmma_bf16(ap0, bv0, o_acc[n]);
            o_acc[n] = wmma_bf16(ap1, bv1, o_acc[n]);
        }
        stage_wait();
        __syncthreads();
    }

    // Normalize and store ctx in (B,S, h*64+dk) bf16 layout.
#pragma unroll
    for (int r = 0; r < 8; ++r) {
        float inv = 1.0f / fmaxf(l_run[r], 1e-20f);
        int qg = q0 + row_off + r;
        size_t obase = ((size_t)(b * SEQ + qg)) * D_MODEL + h * D_K;
#pragma unroll
        for (int n = 0; n < 4; ++n)
            ctx[obase + n * 16 + l16] = (bf16_t)(o_acc[n][r] * inv);
    }
}

// ---------------------------------------------------------------------------
// Kernel 3: output projection, double-buffered async staging.
// grid = (ROWS/128, D/64); block 128 (32x64 per wave). Wo^T prepacked bf16.
// ---------------------------------------------------------------------------
__global__ __launch_bounds__(128) void out_proj_kernel(
    const bf16_t* __restrict__ ctx, const bf16_t* __restrict__ Wot,
    const float* __restrict__ bo, float* __restrict__ out)
{
    __shared__ __align__(32) bf16_t As[2][128][32];
    __shared__ __align__(32) bf16_t Bs[2][32][64];

    const int t = threadIdx.x;
    const int wave = t >> 5, lane = t & 31, l16 = lane & 15, hi = lane >> 4;
    const int m0 = blockIdx.x * 128;
    const int n0 = blockIdx.y * 64;

    auto stage = [&](int buf, int kkc) {
#pragma unroll
        for (int i = 0; i < 4; ++i) {
            int u = t + i * 128;
            int row = u >> 2, c8 = (u & 3) * 8;
            stage16(&As[buf][row][c8],
                    ctx + (size_t)(m0 + row) * D_MODEL + kkc + c8);
        }
#pragma unroll
        for (int i = 0; i < 2; ++i) {
            int u = t + i * 128;
            int kr = u >> 3, c8 = (u & 7) * 8;
            stage16(&Bs[buf][kr][c8],
                    Wot + (size_t)(kkc + kr) * D_MODEL + n0 + c8);
        }
    };

    v8f acc[2][4];
#pragma unroll
    for (int ms = 0; ms < 2; ++ms)
#pragma unroll
        for (int n = 0; n < 4; ++n)
#pragma unroll
            for (int r = 0; r < 8; ++r) acc[ms][n][r] = 0.0f;

    stage(0, 0);
    stage_wait();
    __syncthreads();

    for (int kk = 0; kk < D_MODEL; kk += 32) {
        const int cur = (kk >> 5) & 1;
        if (kk + 32 < D_MODEL) stage(cur ^ 1, kk + 32);

        v16bf a[2];
#pragma unroll
        for (int ms = 0; ms < 2; ++ms) {
            const bf16_t* ar = &As[cur][wave * 32 + ms * 16 + l16][0];
            a[ms] = load_frag(ar + hi * 8, ar + 16 + hi * 8);
        }
#pragma unroll
        for (int n = 0; n < 4; ++n) {
            v16bf b = load_bcol(&Bs[cur][0][n * 16 + l16], 64, hi);
#pragma unroll
            for (int ms = 0; ms < 2; ++ms)
                acc[ms][n] = wmma_bf16(a[ms], b, acc[ms][n]);
        }
        stage_wait();
        __syncthreads();
    }

#pragma unroll
    for (int ms = 0; ms < 2; ++ms)
#pragma unroll
        for (int n = 0; n < 4; ++n) {
            int col = n0 + n * 16 + l16;
            float bias = bo[col];
#pragma unroll
            for (int r = 0; r < 8; ++r) {
                int row = m0 + wave * 32 + ms * 16 + (hi ? 8 : 0) + r;
                out[(size_t)row * D_MODEL + col] = acc[ms][n][r] + bias;
            }
        }
}

// ---------------------------------------------------------------------------
extern "C" void kernel_launch(void* const* d_in, const int* in_sizes, int n_in,
                              void* d_out, int out_size, void* d_ws, size_t ws_size,
                              hipStream_t stream)
{
    (void)in_sizes; (void)n_in; (void)out_size; (void)ws_size;

    const float* x    = (const float*)d_in[0];
    const int*   mask = (const int*)  d_in[1];
    const float* Wq   = (const float*)d_in[2];
    const float* Wk   = (const float*)d_in[3];
    const float* Wv   = (const float*)d_in[4];
    const float* Wo   = (const float*)d_in[5];
    const float* bo   = (const float*)d_in[6];
    float* out        = (float*)d_out;

    const size_t NELEM = (size_t)ROWS * D_MODEL;        // 8M
    const size_t WELEM = (size_t)D_MODEL * D_MODEL;     // 1M
    bf16_t* xb  = (bf16_t*)d_ws;          // 16 MB
    bf16_t* wt  = xb  + NELEM;            //  8 MB (4 transposed weights)
    bf16_t* qb  = wt  + 4 * WELEM;        // 16 MB
    bf16_t* kt  = qb  + NELEM;            // 16 MB (K transposed: bh, dk, s)
    bf16_t* vb  = kt  + NELEM;            // 16 MB
    bf16_t* ctx = vb  + NELEM;            // 16 MB  (total 88 MB)

    convert_x_kernel<<<dim3(ROWS * D_MODEL / 1024), 256, 0, stream>>>(x, xb);
    transpose_w_kernel<<<dim3(D_MODEL / 64, D_MODEL / 64, 4), 128, 0, stream>>>(
        Wq, Wk, Wv, Wo, wt);
    qkv_proj_kernel<<<dim3(ROWS / 128, D_MODEL / 64, 3), 128, 0, stream>>>(
        xb, wt, qb, kt, vb);
    flash_attn_kernel<<<dim3(SEQ / 64, BATCH * N_HEADS), 128, 0, stream>>>(
        qb, kt, vb, mask, ctx);
    out_proj_kernel<<<dim3(ROWS / 128, D_MODEL / 64), 128, 0, stream>>>(
        ctx, wt + 3 * WELEM, bo, out);
}
